// Attention_30305289240928
// MI455X (gfx1250) — matrix operand order repaired and checked by reading, not compiled, not run on hardware
//
#include <hip/hip_runtime.h>
#include <hip/hip_bf16.h>

// ---------------------------------------------------------------------------
// Problem constants (match reference)
// ---------------------------------------------------------------------------
constexpr int SEQ  = 2048;
constexpr int HID  = 2048;
constexpr int NH   = 16;
constexpr int HD   = 128;          // head dim
constexpr int QKVW = 3 * HID;      // 6144, fused QKV row width
constexpr float EPS = 1e-5f;

typedef __attribute__((ext_vector_type(16))) _Float16 v16h;
typedef __attribute__((ext_vector_type(8)))  float    v8f;

// Optional gfx1250 async global->LDS copy path (falls back to plain copies).
#if defined(__has_builtin)
#if __has_builtin(__builtin_amdgcn_global_load_async_to_lds_b128) && \
    __has_builtin(__builtin_amdgcn_s_wait_asynccnt)
#define HAVE_ASYNC_LDS 1
#endif
#if __has_builtin(__builtin_amdgcn_mov_dpp8) && \
    __has_builtin(__builtin_amdgcn_update_dpp)
#define HAVE_DPP 1
#endif
#endif
#ifndef HAVE_ASYNC_LDS
#define HAVE_ASYNC_LDS 0
#endif
#ifndef HAVE_DPP
#define HAVE_DPP 0
#endif

#if HAVE_ASYNC_LDS
// Builtin expects pointers to GCC-style 4 x i32 vectors in AS1 (global) / AS3 (LDS).
typedef int v4ia __attribute__((vector_size(16)));
typedef __attribute__((address_space(1))) v4ia* async_gptr;
typedef __attribute__((address_space(3))) v4ia* async_lptr;
#endif

// ---------------------------------------------------------------------------
// VALU-only 16-lane reductions (DPP butterflies; no LDS round trips).
// ---------------------------------------------------------------------------
__device__ __forceinline__ int   f2i(float f) { union { float f; int i; } u; u.f = f; return u.i; }
__device__ __forceinline__ float i2f(int i)   { union { float f; int i; } u; u.i = i; return u.f; }

#if HAVE_DPP
// dpp8 lane selects for xor-1/2/4 within groups of 8 lanes.
#define DPP8_SEL(a,b,c,d,e,f,gg,hh) \
    ((a)|((b)<<3)|((c)<<6)|((d)<<9)|((e)<<12)|((f)<<15)|((gg)<<18)|((hh)<<21))
template <unsigned SEL>
__device__ __forceinline__ float dpp8f(float x) {
    return i2f(__builtin_amdgcn_mov_dpp8(f2i(x), SEL));
}
__device__ __forceinline__ float row_ror8(float x) {
    // DPP16 row_ror:8 == xor-8 within a 16-lane row.
    return i2f(__builtin_amdgcn_update_dpp(0, f2i(x), 0x128, 0xf, 0xf, true));
}
__device__ __forceinline__ float hmax16(float v) {
    v = fmaxf(v, dpp8f<DPP8_SEL(1,0,3,2,5,4,7,6)>(v));   // xor 1
    v = fmaxf(v, dpp8f<DPP8_SEL(2,3,0,1,6,7,4,5)>(v));   // xor 2
    v = fmaxf(v, dpp8f<DPP8_SEL(4,5,6,7,0,1,2,3)>(v));   // xor 4
    v = fmaxf(v, row_ror8(v));                           // cross 8
    return v;
}
__device__ __forceinline__ float hsum16(float v) {
    v += dpp8f<DPP8_SEL(1,0,3,2,5,4,7,6)>(v);
    v += dpp8f<DPP8_SEL(2,3,0,1,6,7,4,5)>(v);
    v += dpp8f<DPP8_SEL(4,5,6,7,0,1,2,3)>(v);
    v += row_ror8(v);
    return v;
}
#else
__device__ __forceinline__ float hmax16(float v) {
    for (int off = 1; off < 16; off <<= 1) v = fmaxf(v, __shfl_xor(v, off, 32));
    return v;
}
__device__ __forceinline__ float hsum16(float v) {
    for (int off = 1; off < 16; off <<= 1) v += __shfl_xor(v, off, 32);
    return v;
}
#endif

// WMMA operand fragment viewed multiple ways.
union Frag {
    v16h h;
    unsigned int u[8];
    uint4 q[2];
    _Float16 e[16];
};

// A-matrix (16x32 f16, ISA 7.12.2): lane group g holds halves [8g..8g+7] in
// VGPRs 0-3 and [16+8g..16+8g+7] in VGPRs 4-7 -> two aligned B128 loads.
__device__ __forceinline__ Frag loadFragA(const _Float16* base, int g) {
    Frag f;
    f.q[0] = *(const uint4*)(base + (g << 3));
    f.q[1] = *(const uint4*)(base + (g << 3) + 16);
    return f;
}
// B-matrix (32x16 f16): lane group g holds halves [8g..8g+7] and [16+8g..16+8g+7] -> two B128.
__device__ __forceinline__ Frag loadFragB(const _Float16* base, int g) {
    Frag f;
    f.q[0] = *(const uint4*)(base + (g << 3));
    f.q[1] = *(const uint4*)(base + (g << 3) + 16);
    return f;
}

// Copy 8 x B128 from global to LDS (async if available).
__device__ __forceinline__ void copy8x16(const uint4* gsrc, uint4* ldst) {
#if HAVE_ASYNC_LDS
    #pragma unroll
    for (int i = 0; i < 8; ++i)
        __builtin_amdgcn_global_load_async_to_lds_b128(
            (async_gptr)(gsrc + i), (async_lptr)(ldst + i), 0, 0);
#else
    #pragma unroll
    for (int i = 0; i < 8; ++i) ldst[i] = gsrc[i];
#endif
}
__device__ __forceinline__ void wait_staging() {
#if HAVE_ASYNC_LDS
    __builtin_amdgcn_s_wait_asynccnt(0);
#endif
}

// ---------------------------------------------------------------------------
// Kernel 1: RMSNorm (no weight, per reference) -> f16 activations
// ---------------------------------------------------------------------------
__global__ void __launch_bounds__(256, 1)
rmsnorm_f16(const float* __restrict__ x, _Float16* __restrict__ xn) {
    __shared__ float red[256];
    const int row = blockIdx.x;
    const int tid = threadIdx.x;
    const float* xr = x + (long)row * HID;
    float ss = 0.0f;
    for (int c = tid; c < HID; c += 256) {
        float v = xr[c];
        ss += v * v;
    }
    red[tid] = ss;
    __syncthreads();
    for (int s = 128; s > 0; s >>= 1) {
        if (tid < s) red[tid] += red[tid + s];
        __syncthreads();
    }
    const float inv = rsqrtf(red[0] * (1.0f / (float)HID) + EPS);
    _Float16* xo = xn + (long)row * HID;
    for (int c = tid; c < HID; c += 256) {
        xo[c] = (_Float16)(xr[c] * inv);
    }
}

// ---------------------------------------------------------------------------
// Kernel 2/4: C[M,N] = A[M,K](f16) * W[N,K]^T (W fp32, converted in-tile).
// Block tile 128x128, BK=32, 8 waves each owning 32x64 (2x4 WMMA frags).
// ---------------------------------------------------------------------------
template <typename OutT>
__global__ void __launch_bounds__(256, 1)
gemm_xwT(const _Float16* __restrict__ A,
         const float* __restrict__ W,
         OutT* __restrict__ C,
         int M, int N, int K) {
    constexpr int LDA = 40;   // halves; 80B rows (16B aligned, bank-rotated)
    __shared__ _Float16 As[128 * LDA];
    __shared__ _Float16 Bs[128 * LDA];

    const int tid  = threadIdx.x;
    const int wave = tid >> 5;
    const int lane = tid & 31;
    const int g    = lane >> 4;
    const int lr   = lane & 15;
    const int wm   = wave >> 1;   // 0..3
    const int wn   = wave & 1;    // 0..1

    const long bm0 = (long)blockIdx.y * 128;
    const long bn0 = (long)blockIdx.x * 128;

    v8f acc[2][4];
    #pragma unroll
    for (int i = 0; i < 2; ++i)
        #pragma unroll
        for (int j = 0; j < 4; ++j)
            #pragma unroll
            for (int r = 0; r < 8; ++r)
                acc[i][j][r] = 0.0f;

    const int ldr = tid >> 1;          // 0..127
    const int ldc = (tid & 1) << 4;    // 0 or 16

    for (int k0 = 0; k0 < K; k0 += 32) {
        // ---- stage A tile: 16 halves per thread via 2x B128 ----
        const uint4* asrc = (const uint4*)(A + (bm0 + ldr) * (long)K + k0 + ldc);
        uint4* adst = (uint4*)&As[ldr * LDA + ldc];
        adst[0] = asrc[0];
        adst[1] = asrc[1];
        // ---- stage W tile: 16 floats -> 16 halves, packed B128 stores ----
        const float4* wsrc = (const float4*)(W + (bn0 + ldr) * (long)K + k0 + ldc);
        union { _Float16 h[16]; uint4 q[2]; } tmp;
        #pragma unroll
        for (int i = 0; i < 4; ++i) {
            float4 f = wsrc[i];
            tmp.h[4 * i + 0] = (_Float16)f.x;
            tmp.h[4 * i + 1] = (_Float16)f.y;
            tmp.h[4 * i + 2] = (_Float16)f.z;
            tmp.h[4 * i + 3] = (_Float16)f.w;
        }
        uint4* bdst = (uint4*)&Bs[ldr * LDA + ldc];
        bdst[0] = tmp.q[0];
        bdst[1] = tmp.q[1];
        if (k0 + 32 < K) {   // prefetch next K-tile while WMMAs run
            __builtin_prefetch((const void*)(asrc + 4), 0, 1);
            __builtin_prefetch((const void*)(wsrc + 8), 0, 1);
        }
        __syncthreads();

        Frag af[2], bf[4];
        #pragma unroll
        for (int fm = 0; fm < 2; ++fm)
            af[fm] = loadFragA(&As[(wm * 32 + fm * 16 + lr) * LDA], g);
        #pragma unroll
        for (int fn = 0; fn < 4; ++fn)
            bf[fn] = loadFragB(&Bs[(wn * 64 + fn * 16 + lr) * LDA], g);
        #pragma unroll
        for (int fm = 0; fm < 2; ++fm)
            #pragma unroll
            for (int fn = 0; fn < 4; ++fn)
                acc[fm][fn] = __builtin_amdgcn_wmma_f32_16x16x32_f16(
                    false, af[fm].h, false, bf[fn].h,
                    (short)0, acc[fm][fn], false, false);
        __syncthreads();
    }

    // C layout: VGPR r -> row (r + 8g), lane lr -> col.
    #pragma unroll
    for (int fm = 0; fm < 2; ++fm)
        #pragma unroll
        for (int fn = 0; fn < 4; ++fn)
            #pragma unroll
            for (int r = 0; r < 8; ++r) {
                const long row = bm0 + wm * 32 + fm * 16 + r + 8 * g;
                const long col = bn0 + wn * 64 + fn * 16 + lr;
                C[row * (long)N + col] = (OutT)acc[fm][fn][r];
            }
}

// ---------------------------------------------------------------------------
// Kernel 3: causal flash attention over the fused QKV buffer (f16).
// Grid: (head, query-block of 128). 8 waves, each owns 16 query rows.
// Key blocks processed in two 64-wide halves to cap register pressure.
// Dynamic LDS: Qs + Ks + Vt (transposed V) + Pl, each [128][136] halves.
// ---------------------------------------------------------------------------
__global__ void __launch_bounds__(256, 1)
flash_attn(const _Float16* __restrict__ qkv,
           _Float16* __restrict__ out) {
    constexpr int LDP = 136;                 // halves; rows 272B (16B aligned)
    extern __shared__ _Float16 smem[];
    _Float16* Qs = smem;                     // [128][LDP] query tile, row = q
    _Float16* Ks = smem + 128 * LDP;         // [128][LDP] key tile,   row = t
    _Float16* Vt = smem + 2 * 128 * LDP;     // [128][LDP] value tile, row = d
    _Float16* Pl = smem + 3 * 128 * LDP;     // [128][LDP] probs,      row = q

    const int h    = blockIdx.x;
    const int qb   = blockIdx.y;
    const int tid  = threadIdx.x;
    const int wave = tid >> 5;
    const int lane = tid & 31;
    const int g    = lane >> 4;
    const int lr   = lane & 15;

    const long RS   = QKVW;               // 6144 halves per token row
    const long hofQ = (long)h * 384;
    const long hofK = hofQ + 128;
    const long hofV = hofQ + 256;
    const int  qrow0 = qb * 128 + wave * 16;   // this wave's first query row
    const float scale = 0.08838834764831845f;  // 1/sqrt(128)

    // Staging decomposition (256 threads):
    const int str = tid >> 1;            // Q/K tile: row 0..127
    const int stc = (tid & 1) << 6;      //           col 0 or 64
    const int vtp = (tid & 63) << 1;     // V tile: t pair base (even)
    const int vdb = (tid >> 6) << 5;     //         d base 0/32/64/96

    // ---- stage Q tile once (visible after first in-loop barriers) ----
    copy8x16((const uint4*)(qkv + (long)(qb * 128 + str) * RS + hofQ + stc),
             (uint4*)&Qs[str * LDP + stc]);

    float m_i[8], l_i[8];
    v8f o[8];
    #pragma unroll
    for (int r = 0; r < 8; ++r) { m_i[r] = -3.0e38f; l_i[r] = 0.0f; }
    #pragma unroll
    for (int nd = 0; nd < 8; ++nd)
        #pragma unroll
        for (int r = 0; r < 8; ++r) o[nd][r] = 0.0f;

    for (int j = 0; j <= qb; ++j) {
        const int kb0 = j * 128;

        __syncthreads();   // previous iteration's LDS reads complete
        // ---- stage K tile (row-major) ----
        copy8x16((const uint4*)(qkv + (long)(kb0 + str) * RS + hofK + stc),
                 (uint4*)&Ks[str * LDP + stc]);
        // ---- stage V tile transposed: Vt[d][t], packed b32 LDS stores ----
        {
            union Row { uint4 q[4]; unsigned short s[32]; } r0, r1;
            const uint4* v0 = (const uint4*)(qkv + (long)(kb0 + vtp) * RS + hofV + vdb);
            const uint4* v1 = (const uint4*)(qkv + (long)(kb0 + vtp + 1) * RS + hofV + vdb);
            #pragma unroll
            for (int i = 0; i < 4; ++i) { r0.q[i] = v0[i]; r1.q[i] = v1[i]; }
            #pragma unroll
            for (int dd = 0; dd < 32; ++dd)
                *(unsigned int*)&Vt[(vdb + dd) * LDP + vtp] =
                    (unsigned int)r0.s[dd] | ((unsigned int)r1.s[dd] << 16);
        }
        if (j < qb) {   // prefetch next key block
            __builtin_prefetch((const void*)(qkv + (long)(kb0 + 128 + str) * RS + hofK + stc), 0, 1);
            __builtin_prefetch((const void*)(qkv + (long)(kb0 + 128 + str) * RS + hofV + stc), 0, 1);
        }
        wait_staging();
        __syncthreads();   // tiles visible

        #pragma unroll
        for (int half = 0; half < 2; ++half) {
            const int c0 = half << 6;           // key column base in tile
            // Skip fully-masked upper half of diagonal block (wave-uniform).
            if (j == qb && half == 1 && wave < 4) continue;

            // ---- S = Q K^T * scale (pipelined: load frag n+1 before wmma n)
            v8f sf[4];
            #pragma unroll
            for (int nf = 0; nf < 4; ++nf)
                #pragma unroll
                for (int r = 0; r < 8; ++r) sf[nf][r] = 0.0f;
            #pragma unroll
            for (int kk = 0; kk < 4; ++kk) {
                const Frag qf = loadFragA(&Qs[(wave * 16 + lr) * LDP + kk * 32], g);
                Frag kc = loadFragB(&Ks[(c0 + lr) * LDP + kk * 32], g);
                #pragma unroll
                for (int nf = 0; nf < 4; ++nf) {
                    Frag kn;
                    if (nf < 3)
                        kn = loadFragB(&Ks[(c0 + (nf + 1) * 16 + lr) * LDP + kk * 32], g);
                    sf[nf] = __builtin_amdgcn_wmma_f32_16x16x32_f16(
                                 false, qf.h, false, kc.h, (short)0, sf[nf], false, false);
                    if (nf < 3) kc = kn;
                }
            }
            #pragma unroll
            for (int nf = 0; nf < 4; ++nf)
                #pragma unroll
                for (int r = 0; r < 8; ++r) sf[nf][r] *= scale;

            // ---- causal mask on the diagonal block ----
            if (j == qb) {
                #pragma unroll
                for (int nf = 0; nf < 4; ++nf) {
                    const int tcol = kb0 + c0 + nf * 16 + lr;
                    #pragma unroll
                    for (int r = 0; r < 8; ++r) {
                        const int srow = qrow0 + r + 8 * g;
                        if (tcol > srow) sf[nf][r] = -3.0e38f;
                    }
                }
            }

            // ---- online softmax (DPP butterfly reductions, 16-lane rows) ---
            float alpha[8];
            #pragma unroll
            for (int r = 0; r < 8; ++r) {
                float mx = m_i[r];
                #pragma unroll
                for (int nf = 0; nf < 4; ++nf) mx = fmaxf(mx, sf[nf][r]);
                mx = hmax16(mx);
                alpha[r] = __expf(m_i[r] - mx);
                m_i[r] = mx;
            }
            float rs[8];
            #pragma unroll
            for (int r = 0; r < 8; ++r) rs[r] = 0.0f;
            #pragma unroll
            for (int nf = 0; nf < 4; ++nf)
                #pragma unroll
                for (int r = 0; r < 8; ++r) {
                    const float p = __expf(sf[nf][r] - m_i[r]);
                    rs[r] += p;
                    Pl[(wave * 16 + r + 8 * g) * LDP + c0 + nf * 16 + lr] = (_Float16)p;
                }
            #pragma unroll
            for (int r = 0; r < 8; ++r) {
                l_i[r] = l_i[r] * alpha[r] + hsum16(rs[r]);
            }
            #pragma unroll
            for (int nd = 0; nd < 8; ++nd)
                #pragma unroll
                for (int r = 0; r < 8; ++r) o[nd][r] *= alpha[r];

            // ---- O += P V (pipelined; per-wave LDS region, DS in order) ----
            #pragma unroll
            for (int kk = 0; kk < 2; ++kk) {
                const Frag pf = loadFragA(&Pl[(wave * 16 + lr) * LDP + c0 + kk * 32], g);
                Frag vc = loadFragB(&Vt[lr * LDP + c0 + kk * 32], g);
                #pragma unroll
                for (int nd = 0; nd < 8; ++nd) {
                    Frag vn;
                    if (nd < 7)
                        vn = loadFragB(&Vt[((nd + 1) * 16 + lr) * LDP + c0 + kk * 32], g);
                    o[nd] = __builtin_amdgcn_wmma_f32_16x16x32_f16(
                                false, pf.h, false, vc.h, (short)0, o[nd], false, false);
                    if (nd < 7) vc = vn;
                }
            }
        }
    }

    // ---- finalize: O / l, write f16 (S, nh*hd) ----
    #pragma unroll
    for (int nd = 0; nd < 8; ++nd)
        #pragma unroll
        for (int r = 0; r < 8; ++r) {
            const long s = qrow0 + r + 8 * g;
            const int  d = nd * 16 + lr;
            out[s * (long)HID + h * HD + d] = (_Float16)(o[nd][r] / l_i[r]);
        }
}

// ---------------------------------------------------------------------------
// Launch: rmsnorm -> QKV gemm -> flash attention -> out-proj gemm
// ---------------------------------------------------------------------------
extern "C" void kernel_launch(void* const* d_in, const int* in_sizes, int n_in,
                              void* d_out, int out_size, void* d_ws, size_t ws_size,
                              hipStream_t stream) {
    (void)in_sizes; (void)n_in; (void)out_size; (void)ws_size;
    const float* x    = (const float*)d_in[0];
    const float* wqkv = (const float*)d_in[1];
    const float* wo   = (const float*)d_in[2];
    float*       out  = (float*)d_out;

    char* ws = (char*)d_ws;
    _Float16* xn   = (_Float16*)ws;                                  // 8 MB
    _Float16* qkv  = (_Float16*)(ws + (size_t)SEQ * HID * 2);        // 24 MB
    _Float16* attn = (_Float16*)(ws + (size_t)SEQ * HID * 2
                                    + (size_t)SEQ * QKVW * 2);       // 8 MB

    rmsnorm_f16<<<SEQ, 256, 0, stream>>>(x, xn);
    gemm_xwT<_Float16><<<dim3(QKVW / 128, SEQ / 128), 256, 0, stream>>>(
        xn, wqkv, qkv, SEQ, QKVW, HID);
    const size_t attn_lds = (size_t)4 * 128 * 136 * sizeof(_Float16);  // 139264 B
    flash_attn<<<dim3(NH, SEQ / 128), 256, attn_lds, stream>>>(qkv, attn);
    gemm_xwT<float><<<dim3(HID / 128, SEQ / 128), 256, 0, stream>>>(
        attn, wo, out, SEQ, HID, HID);
}
